// CoverageLoss_49143015801515
// MI455X (gfx1250) — compile-verified, matrix operand order checked by reading, not run on hardware
//
#include <hip/hip_runtime.h>
#include <cstdint>
#include <cstddef>

// ---------------------------------------------------------------------------
// CDNA5 / gfx1250 implementation of the coverage+size loss.
//   - v_wmma_f32_16x16x32_f16 performs the e-reduction of |sample - latent|
//     (B = ones -> K-layout of A irrelevant; f16 inputs, f32 accumulate)
//   - latent tiles double-buffered in LDS via global_load_async_to_lds_b128
//     (ASYNCcnt pipeline: stage next tile while computing current)
//   - abs done in f16 domain: 1x v_and 0x7fff7fff per 2 elements
//   - top-4 maintenance with v_med3_f32 (4 ops/candidate)
// ---------------------------------------------------------------------------

typedef __attribute__((ext_vector_type(16))) _Float16 v16h;
typedef __attribute__((ext_vector_type(8)))  float    v8f;
typedef __attribute__((ext_vector_type(4)))  float    v4f;

__device__ __forceinline__ void async_ld_b128(uint32_t lds_off, const void* gaddr) {
  asm volatile("global_load_async_to_lds_b128 %0, %1, off"
               :: "v"(lds_off), "v"(gaddr) : "memory");
}
__device__ __forceinline__ void wait_asynccnt0() {
  asm volatile("s_wait_asynccnt 0" ::: "memory");
}

// keep 4 smallest (ascending t0<=t1<=t2<=t3); insert = clamp = med3
__device__ __forceinline__ void ins4(float t[4], float x) {
  float s0 = fminf(t[0], x);
  float s1 = __builtin_amdgcn_fmed3f(x, t[0], t[1]);
  float s2 = __builtin_amdgcn_fmed3f(x, t[1], t[2]);
  float s3 = __builtin_amdgcn_fmed3f(x, t[2], t[3]);
  t[0] = s0; t[1] = s1; t[2] = s2; t[3] = s3;
}

// ---------------------------------------------------------------------------
// size loss: sum over rows of relu(||row||_1 - size)^2 into *acc
// ---------------------------------------------------------------------------
template<int E>
__global__ __launch_bounds__(256) void size_kernel(const float* __restrict__ x,
                                                   float size, float* acc, int rows) {
  const int lane = threadIdx.x & 31;
  const int row  = blockIdx.x * 8 + (threadIdx.x >> 5);
  if (row >= rows) return;
  const float* r = x + (size_t)row * E;
  float s = 0.f;
  #pragma unroll
  for (int e = lane; e < E; e += 32) s += __builtin_fabsf(r[e]);
  #pragma unroll
  for (int m = 16; m; m >>= 1) s += __shfl_xor(s, m, 32);
  if (lane == 0) {
    float v = s - size; v = v > 0.f ? v : 0.f;
    atomicAdd(acc, v * v);
  }
}

// ---------------------------------------------------------------------------
// coverage kernel: per sample, 4 smallest L1 distances to all N latents.
//   TAIL (PUSH=false): tail_out[s] = mean of 4 smallest
//   PUSH (PUSH=true) : atomicAdd(acc, sum of squares of 4 smallest),
//                      samples gathered through far_idx
// Block = 8 waves, each wave owns SPW=2 samples; block streams 64-latent
// tiles through double-buffered LDS via async loads. WMMA with B=ones
// reduces the 32-wide e-chunks: D[m,n] = sum_k |diff|[m,k].
// ---------------------------------------------------------------------------
template<int E, bool PUSH>
__global__ __launch_bounds__(256)
void cov_kernel(const float* __restrict__ samples, const float* __restrict__ latents,
                const int* __restrict__ far_idx, float* tail_out, float* acc,
                int nSamples, int N) {
  constexpr int CH     = E / 32;   // 32-wide e-chunks per WMMA chain
  constexpr int STRIDE = E + 4;    // padded LDS row stride -> conflict-free b128 reads
  constexpr int TL     = 64;       // latents per staged tile
  constexpr int SPW    = 2;        // samples per wave

  __shared__ __align__(16) float lat_s[2][TL * STRIDE];

  const int tid   = threadIdx.x;
  const int lane  = tid & 31;
  const int wave  = tid >> 5;
  const int sbase = (blockIdx.x * 8 + wave) * SPW;

  // B = all ones (f16): any K-permutation of A sums correctly.
  v16h ones;
  #pragma unroll
  for (int i = 0; i < 16; ++i) ones[i] = (_Float16)1.0f;

  // per-lane sample registers: 16 f32 per (sample, chunk); lanes<16 take the
  // low 16 e's of each 32-chunk, lanes>=16 the high 16 e's.
  float samp[SPW][CH][16];
  #pragma unroll
  for (int s = 0; s < SPW; ++s) {
    int gs = sbase + s;
    const int si = PUSH ? far_idx[gs] : gs;
    const float* sp = samples + (size_t)si * E;
    #pragma unroll
    for (int c = 0; c < CH; ++c) {
      const v4f* p = (const v4f*)(sp + c * 32 + ((lane < 16) ? 0 : 16));
      #pragma unroll
      for (int q = 0; q < 4; ++q) {
        v4f v = p[q];
        #pragma unroll
        for (int k = 0; k < 4; ++k) samp[s][c][4 * q + k] = v[k];
      }
    }
  }

  float t[SPW][4];
  #pragma unroll
  for (int s = 0; s < SPW; ++s)
    #pragma unroll
    for (int j = 0; j < 4; ++j) t[s][j] = __builtin_inff();

  const int b128PerRow = E / 4;
  const int totB = TL * b128PerRow;

  auto stage = [&](int buf, int tile0) {
    for (int idx = tid; idx < totB; idx += 256) {
      const int row = idx / b128PerRow;
      const int c4  = idx % b128PerRow;
      uint32_t loff  = (uint32_t)(size_t)&lat_s[buf][row * STRIDE + c4 * 4];
      const float* g = latents + (size_t)(tile0 + row) * E + c4 * 4;
      async_ld_b128(loff, g);
    }
  };

  // prologue: stage tile 0
  stage(0, 0);
  wait_asynccnt0();
  __syncthreads();

  const int nTiles = N / TL;
  for (int tt = 0; tt < nTiles; ++tt) {
    const int buf = tt & 1;
    if (tt + 1 < nTiles) stage(buf ^ 1, (tt + 1) * TL);  // overlap DMA with compute

    #pragma unroll
    for (int st = 0; st < TL / 16; ++st) {
      const int lrow = st * 16 + (lane & 15);
      float latv[CH][16];
      #pragma unroll
      for (int c = 0; c < CH; ++c) {
        const v4f* lp = (const v4f*)&lat_s[buf][lrow * STRIDE + c * 32 + ((lane < 16) ? 0 : 16)];
        #pragma unroll
        for (int q = 0; q < 4; ++q) {
          v4f v = lp[q];
          #pragma unroll
          for (int k = 0; k < 4; ++k) latv[c][4 * q + k] = v[k];
        }
      }
      #pragma unroll
      for (int s = 0; s < SPW; ++s) {
        v8f d = {};
        #pragma unroll
        for (int c = 0; c < CH; ++c) {
          // signed diff -> f16 (v_pk_add neg + v_cvt_pk), abs in f16 domain
          union { v16h h; uint32_t u[8]; } a;
          #pragma unroll
          for (int i = 0; i < 16; ++i)
            a.h[i] = (_Float16)(samp[s][c][i] - latv[c][i]);
          #pragma unroll
          for (int i = 0; i < 8; ++i) a.u[i] &= 0x7fff7fffu;
          d = __builtin_amdgcn_wmma_f32_16x16x32_f16(
                false, a.h, false, ones, (short)0, d, false, false);
        }
        // lanes<16 see latents M0-7, lanes>=16 see M8-15 (disjoint halves)
        #pragma unroll
        for (int j = 0; j < 8; ++j) ins4(t[s], d[j]);
      }
    }

    wait_asynccnt0();   // next tile's DMA complete (this wave)
    __syncthreads();    // all waves done with compute + DMA
  }

  // merge the two half-wave top-4 lists, then emit
  #pragma unroll
  for (int s = 0; s < SPW; ++s) {
    float o0 = __shfl_xor(t[s][0], 16, 32);
    float o1 = __shfl_xor(t[s][1], 16, 32);
    float o2 = __shfl_xor(t[s][2], 16, 32);
    float o3 = __shfl_xor(t[s][3], 16, 32);
    ins4(t[s], o0); ins4(t[s], o1); ins4(t[s], o2); ins4(t[s], o3);
    if (lane == 0) {
      const int gs = sbase + s;
      if (PUSH) {
        float v = t[s][0]*t[s][0] + t[s][1]*t[s][1] + t[s][2]*t[s][2] + t[s][3]*t[s][3];
        atomicAdd(acc, v);
      } else {
        tail_out[gs] = 0.25f * (t[s][0] + t[s][1] + t[s][2] + t[s][3]);
      }
    }
  }
}

// ---------------------------------------------------------------------------
// top-64 (largest, ties -> lowest index, matching lax.top_k) over 2048 tails
// ---------------------------------------------------------------------------
__global__ __launch_bounds__(256) void topk_kernel(const float* __restrict__ tail,
                                                   int n, int k, int* out_idx) {
  __shared__ float vals[2048];
  __shared__ float bv[256];
  __shared__ int   bi[256];
  const int tid = threadIdx.x;
  for (int i = tid; i < n; i += 256) vals[i] = tail[i];
  __syncthreads();
  for (int it = 0; it < k; ++it) {
    float best = -__builtin_inff(); int besti = n;
    for (int i = tid; i < n; i += 256) {
      float v = vals[i];
      if (v > best || (v == best && i < besti)) { best = v; besti = i; }
    }
    bv[tid] = best; bi[tid] = besti;
    __syncthreads();
    for (int off = 128; off; off >>= 1) {
      if (tid < off) {
        float v = bv[tid + off]; int j = bi[tid + off];
        if (v > bv[tid] || (v == bv[tid] && j < bi[tid])) { bv[tid] = v; bi[tid] = j; }
      }
      __syncthreads();
    }
    if (tid == 0) { out_idx[it] = bi[0]; vals[bi[0]] = -__builtin_inff(); }
    __syncthreads();
  }
}

__global__ void init_kernel(float* acc) {
  if (threadIdx.x < 4) acc[threadIdx.x] = 0.f;
}

__global__ void fin_kernel(const float* __restrict__ acc, float* __restrict__ out) {
  if (threadIdx.x == 0)
    out[0] = acc[0] * (1.0f / 8192.0f) + acc[1] * (1.0f / 8192.0f)
           + acc[2] * (1.0f / 256.0f)  + acc[3] * (1.0f / 256.0f);
}

// ---------------------------------------------------------------------------
extern "C" void kernel_launch(void* const* d_in, const int* in_sizes, int n_in,
                              void* d_out, int out_size, void* d_ws, size_t ws_size,
                              hipStream_t stream) {
  (void)in_sizes; (void)n_in; (void)out_size; (void)ws_size;
  const float* latS  = (const float*)d_in[0];   // [8192,64]
  const float* latA  = (const float*)d_in[1];   // [8192,32]
  const float* sampS = (const float*)d_in[2];   // [2048,64]
  const float* sampA = (const float*)d_in[3];   // [2048,32]
  float* out = (float*)d_out;

  float* wf    = (float*)d_ws;
  float* acc   = wf;                    // [0]=sizeS [1]=sizeA [2]=covS [3]=covA
  float* tailS = wf + 16;               // 2048
  float* tailA = wf + 16 + 2048;        // 2048
  int*   farS  = (int*)(wf + 16 + 4096);// 64
  int*   farA  = farS + 64;             // 64

  init_kernel<<<1, 32, 0, stream>>>(acc);
  size_kernel<64><<<1024, 256, 0, stream>>>(latS, 1.0f, acc + 0, 8192);
  size_kernel<32><<<1024, 256, 0, stream>>>(latA, 1.0f, acc + 1, 8192);

  // tails: 2048 samples / (8 waves * 2 samples) = 128 blocks
  cov_kernel<64, false><<<128, 256, 0, stream>>>(sampS, latS, nullptr, tailS, nullptr, 2048, 8192);
  cov_kernel<32, false><<<128, 256, 0, stream>>>(sampA, latA, nullptr, tailA, nullptr, 2048, 8192);

  topk_kernel<<<1, 256, 0, stream>>>(tailS, 2048, 64, farS);
  topk_kernel<<<1, 256, 0, stream>>>(tailA, 2048, 64, farA);

  // push loss over the 64 far samples: 64 / 16 = 4 blocks
  cov_kernel<64, true><<<4, 256, 0, stream>>>(sampS, latS, farS, nullptr, acc + 2, 64, 8192);
  cov_kernel<32, true><<<4, 256, 0, stream>>>(sampA, latA, farA, nullptr, acc + 3, 64, 8192);

  fin_kernel<<<1, 32, 0, stream>>>(acc, out);
}
